// GMM_dominik_10831907520640
// MI455X (gfx1250) — matrix operand order repaired and checked by reading, not compiled, not run on hardware
//
#include <hip/hip_runtime.h>
#include <hip/hip_bf16.h>
#include <cstdint>
#include <cstddef>

// Problem constants (match reference)
#define B_    8
#define T_    1024
#define F_    512
#define K_    256
#define L_    16
#define NTOK  (B_ * T_)   // 8192 tokens
#define KD    (2 * F_)    // 1024 augmented K-dim: [x^2 | x]
#define KL    (K_ * L_)   // 4096 rows of W

// GEMM tiling
#define KSTEP   64
#define NSTEP   (KD / KSTEP)       // 16
#define AS_ROW  72                 // padded row stride in halves (36 dwords -> conflict-free)
#define BS_ROW  72
#define AS_TILE (128 * AS_ROW)     // halves per A buffer
#define BS_TILE (64 * BS_ROW)      // halves per B buffer

typedef __attribute__((ext_vector_type(16))) __bf16 v16bf;
typedef __attribute__((ext_vector_type(8)))  float  v8f;

union FragAB { v16bf v; uint4 u[2]; };
union FragC  { v8f   v; float f[8]; };

__device__ __forceinline__ unsigned short f2bf(float f) {
    unsigned int u = __builtin_bit_cast(unsigned int, f);
    u += 0x7FFFu + ((u >> 16) & 1u);
    return (unsigned short)(u >> 16);
}

// ---------------------------------------------------------------------------
// Kernel 1: W rows (bf16) + per-(k,l) partial sums (see round-0 analysis).
// ---------------------------------------------------------------------------
__global__ __launch_bounds__(256)
void k_prep_w(const float* __restrict__ mu, const float* __restrict__ log_cov,
              unsigned short* __restrict__ W, float* __restrict__ slc,
              float* __restrict__ quad) {
    const int wave = threadIdx.x >> 5;
    const int lane = threadIdx.x & 31;
    const int r = blockIdx.x * 8 + wave;          // 0..4095
    const float* lcp = log_cov + (size_t)r * F_;
    const float* mup = mu      + (size_t)r * F_;
    unsigned short* wr = W + (size_t)r * KD;
    float s_lc = 0.f, s_q = 0.f;
    for (int f = lane; f < F_; f += 32) {
        float lc = lcp[f];
        float ic = __expf(-lc);
        float m  = mup[f];
        wr[f]       = f2bf(-0.5f * ic);
        wr[F_ + f]  = f2bf(m * ic);
        s_lc += lc;
        s_q  += m * m * ic;
    }
    for (int d = 16; d; d >>= 1) {
        s_lc += __shfl_xor(s_lc, d, 32);
        s_q  += __shfl_xor(s_q,  d, 32);
    }
    if (lane == 0) { slc[r] = s_lc; quad[r] = s_q; }
}

// ---------------------------------------------------------------------------
// Kernel 2: fused bias const[k,l] = C + log_softmax(log_pi) - 0.5*quad
// ---------------------------------------------------------------------------
__global__ __launch_bounds__(256)
void k_const(const float* __restrict__ log_pi, const float* __restrict__ slc,
             const float* __restrict__ quad, float* __restrict__ cst) {
    const int k = threadIdx.x;                    // 0..255
    const float* lp = log_pi + k * L_;
    float mx = -3.0e38f;
    for (int l = 0; l < L_; ++l) mx = fmaxf(mx, lp[l]);
    float se = 0.f;
    for (int l = 0; l < L_; ++l) se += __expf(lp[l] - mx);
    const float lse   = mx + __logf(se);
    const float Cbase = -470.49652900079243f;     // -256 * ln(2*pi)
    for (int l = 0; l < L_; ++l) {
        int r = k * L_ + l;
        cst[r] = Cbase - 0.5f * slc[r] - 0.5f * quad[r] + (lp[l] - lse);
    }
}

// ---------------------------------------------------------------------------
// Kernel 3: augmented A matrix in bf16:  A[n, :F]=x^2, A[n, F:]=x
// ---------------------------------------------------------------------------
__global__ __launch_bounds__(256)
void k_prep_a(const float* __restrict__ X, unsigned short* __restrict__ A) {
    const int idx = blockIdx.x * 256 + threadIdx.x;   // < NTOK*F_
    const int n = idx >> 9;
    const int f = idx & 511;
    float x = X[idx];
    A[(size_t)n * KD + f]       = f2bf(x * x);
    A[(size_t)n * KD + F_ + f]  = f2bf(x);
}

// ---------------------------------------------------------------------------
// Async global->LDS issue: per thread 4x16B of A tile + 2x16B of B tile.
// GLOBAL_LOAD_ASYNC_TO_LDS adds INST_OFFSET to BOTH the LDS and the global
// address, so consecutive 16B chunks share one LDS-addr and one global-offset
// VGPR. Tracked by ASYNCcnt.
// ---------------------------------------------------------------------------
__device__ __forceinline__ void issue_async(uint32_t aL, uint32_t bL,
                                            uint32_t aG, uint32_t bG,
                                            unsigned long long aBase,
                                            unsigned long long bBase) {
    asm volatile(
        "global_load_async_to_lds_b128 %0, %2, %4 offset:0\n\t"
        "global_load_async_to_lds_b128 %0, %2, %4 offset:16\n\t"
        "global_load_async_to_lds_b128 %0, %2, %4 offset:32\n\t"
        "global_load_async_to_lds_b128 %0, %2, %4 offset:48\n\t"
        "global_load_async_to_lds_b128 %1, %3, %5 offset:0\n\t"
        "global_load_async_to_lds_b128 %1, %3, %5 offset:16"
        :: "v"(aL), "v"(bL), "v"(aG), "v"(bG), "s"(aBase), "s"(bBase)
        : "memory");
}

// Consume one 64-wide K tile from LDS: 2 sub-steps x 4 WMMAs.
__device__ __forceinline__ void compute_tile(const unsigned short* As,
                                             const unsigned short* Bs,
                                             int wm, int wn, int m, int h,
                                             v8f acc[4]) {
#pragma unroll
    for (int sub = 0; sub < KSTEP; sub += 32) {
        // B fragment (32x16): lane (n=m, h) holds K = sub+16h .. sub+16h+15
        FragAB bfr;
        const unsigned short* bp = Bs + (wn * 16 + m) * BS_ROW + sub + 16 * h;
        bfr.u[0] = *(const uint4*)bp;
        bfr.u[1] = *(const uint4*)(bp + 8);
        // A fragments first (distinct regs -> LDS latency pipelined across WMMAs)
        FragAB afr[4];
#pragma unroll
        for (int mt = 0; mt < 4; ++mt) {
            const unsigned short* ap =
                As + (wm * 64 + mt * 16 + m) * AS_ROW + sub + 16 * h;
            afr[mt].u[0] = *(const uint4*)ap;
            afr[mt].u[1] = *(const uint4*)(ap + 8);
        }
#pragma unroll
        for (int mt = 0; mt < 4; ++mt)
            acc[mt] = __builtin_amdgcn_wmma_f32_16x16x32_bf16(
                false, afr[mt].v, false, bfr.v, (short)0, acc[mt], false, false);
    }
}

// ---------------------------------------------------------------------------
// Kernel 4: bf16 WMMA GEMM (8192x4096x1024) + fused bias + logsumexp(L=16).
// Block = 256 threads (8 waves), tile 128(M) x 64(N = 4 classes), K-step 64,
// double-buffered LDS filled by async global->LDS DMA; one barrier per step.
// ---------------------------------------------------------------------------
__global__ __launch_bounds__(256)
void k_gemm(const unsigned short* __restrict__ A,
            const unsigned short* __restrict__ W,
            const float* __restrict__ cst,
            float* __restrict__ out) {
    __shared__ __align__(64) unsigned short As[2 * AS_TILE];  // 36,864 B
    __shared__ __align__(64) unsigned short Bs[2 * BS_TILE];  // 18,432 B

    const int tid  = threadIdx.x;
    const int lane = tid & 31;
    const int wave = tid >> 5;
    const int wm = wave & 1;        // M half (0..1)
    const int wn = wave >> 1;       // class within block (0..3)
    const int m  = lane & 15;       // row (A) / column l (B,C)
    const int h  = lane >> 4;       // half-group

    const int tok0 = blockIdx.x * 128;
    const int cls0 = blockIdx.y * 4;

    // async-load assignments: A 128 rows x 2 col-halves (64B), B 64 rows x 4 (32B)
    const int ar = tid >> 1, ac = (tid & 1) * 32;     // halves
    const int br = tid >> 2, bc = (tid & 3) * 16;     // halves

    // LDS byte offsets (low 32 bits of the flat shared address == LDS offset)
    const uint32_t asBase = (uint32_t)(uintptr_t)&As[0];
    const uint32_t bsBase = (uint32_t)(uintptr_t)&Bs[0];
    const uint32_t aL0 = asBase + (uint32_t)(ar * AS_ROW + ac) * 2u;
    const uint32_t bL0 = bsBase + (uint32_t)(br * BS_ROW + bc) * 2u;
    // global byte offsets from base pointers
    const uint32_t aG0 = (uint32_t)(((tok0 + ar) * KD + ac) * 2);
    const uint32_t bG0 = (uint32_t)(((cls0 * L_ + br) * KD + bc) * 2);
    const unsigned long long aBase = (unsigned long long)(uintptr_t)A;
    const unsigned long long bBase = (unsigned long long)(uintptr_t)W;

    v8f zero = {};
    v8f acc[4];
    acc[0] = zero; acc[1] = zero; acc[2] = zero; acc[3] = zero;

    // prologue: fill buffer 0 with K-tile 0
    issue_async(aL0, bL0, aG0, bG0, aBase, bBase);

    for (int s = 0; s < NSTEP; ++s) {
        const int buf = s & 1;
        asm volatile("s_wait_asynccnt 0x0" ::: "memory"); // my tile landed
        __syncthreads();                                  // all tiles landed,
                                                          // prev compute done
        if (s + 1 < NSTEP) {
            const uint32_t koff = (uint32_t)(s + 1) * (KSTEP * 2);
            issue_async(aL0 + (buf ^ 1) * (AS_TILE * 2),
                        bL0 + (buf ^ 1) * (BS_TILE * 2),
                        aG0 + koff, bG0 + koff, aBase, bBase);
        }
        compute_tile(As + buf * AS_TILE, Bs + buf * BS_TILE, wm, wn, m, h, acc);
    }

    // Epilogue: bias + logsumexp over the 16 columns (= lanes within half-group)
    const int cls = cls0 + wn;
    const float c = cst[cls * L_ + m];     // lane's column l = m
#pragma unroll
    for (int mt = 0; mt < 4; ++mt) {
        FragC u; u.v = acc[mt];
#pragma unroll
        for (int i = 0; i < 8; ++i) {
            float v = u.f[i] + c;
            float mx = v;
            mx = fmaxf(mx, __shfl_xor(mx, 1, 32));
            mx = fmaxf(mx, __shfl_xor(mx, 2, 32));
            mx = fmaxf(mx, __shfl_xor(mx, 4, 32));
            mx = fmaxf(mx, __shfl_xor(mx, 8, 32));
            float e = __expf(v - mx);
            e += __shfl_xor(e, 1, 32);
            e += __shfl_xor(e, 2, 32);
            e += __shfl_xor(e, 4, 32);
            e += __shfl_xor(e, 8, 32);
            if (m == 0) {
                const int token = tok0 + wm * 64 + mt * 16 + i + 8 * h;
                out[(size_t)token * K_ + cls] = mx + __logf(e);
            }
        }
    }
}

// ---------------------------------------------------------------------------
// Workspace layout (needs 24 MiB + 48 KiB):
//   [0, 8 MiB)      W  bf16  (4096 x 1024)
//   [8, 24 MiB)     A  bf16  (8192 x 1024)
//   [24 MiB, ...)   const f32 (4096), slc f32 (4096), quad f32 (4096)
// ---------------------------------------------------------------------------
extern "C" void kernel_launch(void* const* d_in, const int* in_sizes, int n_in,
                              void* d_out, int out_size, void* d_ws, size_t ws_size,
                              hipStream_t stream) {
    (void)in_sizes; (void)n_in; (void)out_size; (void)ws_size;
    const float* X       = (const float*)d_in[0];
    const float* mu      = (const float*)d_in[1];
    const float* log_cov = (const float*)d_in[2];
    const float* log_pi  = (const float*)d_in[3];
    float* out = (float*)d_out;

    char* ws = (char*)d_ws;
    unsigned short* Wb = (unsigned short*)ws;                              // 8 MiB
    unsigned short* Ab = (unsigned short*)(ws + (size_t)8 * 1024 * 1024);  // 16 MiB
    float* cst  = (float*)(ws + (size_t)24 * 1024 * 1024);
    float* slc  = cst + KL;
    float* quad = slc + KL;

    k_prep_w<<<dim3(KL / 8), dim3(256), 0, stream>>>(mu, log_cov, Wb, slc, quad);
    k_const <<<dim3(1),      dim3(256), 0, stream>>>(log_pi, slc, quad, cst);
    k_prep_a<<<dim3((NTOK * F_) / 256), dim3(256), 0, stream>>>(X, Ab);
    k_gemm  <<<dim3(NTOK / 128, K_ / 4), dim3(256), 0, stream>>>(Ab, Wb, cst, out);
}
